// DCN_87952340288138
// MI455X (gfx1250) — matrix operand order repaired and checked by reading, not compiled, not run on hardware
//
#include <hip/hip_runtime.h>

// ---------------- problem dims ----------------
#define BATCH 16384
#define NSP   26
#define EMB   64
#define DDIM  13
#define D0    1677          // NSP*EMB + DDIM
#define D0P   1696          // padded to multiple of 32 (53*32)
#define H1    1024
#define H2    512
#define H3    256
#define VOCAB 50000

typedef unsigned short u16;
typedef unsigned int   u32;
typedef __attribute__((ext_vector_type(4)))  u32     u32x4;
typedef __attribute__((ext_vector_type(4)))  int     v4i;
typedef __attribute__((ext_vector_type(16))) __bf16  v16bf;
typedef __attribute__((ext_vector_type(8)))  float   v8f;

#define AS1 __attribute__((address_space(1)))
#define AS3 __attribute__((address_space(3)))

#if __has_builtin(__builtin_amdgcn_global_load_async_to_lds_b128)
#define ASYNC_LDS 1
#endif

__device__ __forceinline__ u16 f2bf(float f) {
    u32 u = __float_as_uint(f);
    u32 r = u + 0x7FFFu + ((u >> 16) & 1u);   // round-to-nearest-even
    return (u16)(r >> 16);
}

// 16B global -> LDS copy: async DMA path on CDNA5, VGPR round-trip otherwise
__device__ __forceinline__ void cp16(const u16* g, u16* s) {
#if defined(ASYNC_LDS)
    __builtin_amdgcn_global_load_async_to_lds_b128(
        (AS1 v4i*)g, (AS3 v4i*)s, 0, 0);
#else
    *(u32x4*)s = *(const u32x4*)g;
#endif
}

#if defined(ASYNC_LDS)
 #if __has_builtin(__builtin_amdgcn_s_wait_asynccnt)
  #define WAIT_ASYNC(N) __builtin_amdgcn_s_wait_asynccnt(N)
 #else
  #define WAIT_ASYNC(N) asm volatile("s_wait_asynccnt " #N ::: "memory")
 #endif
#else
 #define WAIT_ASYNC(N)
#endif

// ---------------------------------------------------------------------------
// K1: embedding gather -> x0 fp32 (padded), plus per-row dots for the cross
// layer (w_cross[2]) and the front half of the final projection (Wp[:1677]).
// ---------------------------------------------------------------------------
__global__ __launch_bounds__(256) void gather_x0(
    const int*   __restrict__ sparse,
    const float* __restrict__ dense,
    const float* __restrict__ emb,
    const float* __restrict__ wc2,    // w_cross row 2, len D0
    const float* __restrict__ wpf,    // Wp[0:D0]
    float* __restrict__ x0,           // [BATCH][D0P]
    float* __restrict__ s_arr,
    float* __restrict__ p_arr)
{
    int b = blockIdx.x;
    int t = threadIdx.x;
    int idx = sparse[(size_t)b * NSP];     // reference uses column 0 for all tables
    float s = 0.f, p = 0.f;
    for (int e = t; e < D0P; e += 256) {
        float v;
        if (e < NSP * EMB) {
            int tab = e >> 6, off = e & 63;
            v = emb[((size_t)tab * VOCAB + idx) * EMB + off];
        } else if (e < D0) {
            v = dense[(size_t)b * DDIM + (e - NSP * EMB)];
        } else {
            v = 0.f;                       // K padding
        }
        x0[(size_t)b * D0P + e] = v;
        if (e < D0) { s += v * wc2[e]; p += v * wpf[e]; }
    }
    __shared__ float rs[256], rp[256];
    rs[t] = s; rp[t] = p;
    __syncthreads();
    for (int st = 128; st > 0; st >>= 1) {
        if (t < st) { rs[t] += rs[t + st]; rp[t] += rp[t + st]; }
        __syncthreads();
    }
    if (t == 0) { s_arr[b] = rs[0]; p_arr[b] = rp[0]; }
}

// ---------------------------------------------------------------------------
// Column batch-norm statistics: 64 columns per block, 4 row-groups.
// Produces fused scale/shift:  y = x*scale + shift
// ---------------------------------------------------------------------------
__global__ __launch_bounds__(256) void col_stats(
    const float* __restrict__ X, int M, int N, int stride,
    const float* __restrict__ g, const float* __restrict__ be,
    float* __restrict__ scale, float* __restrict__ shift)
{
    int c  = threadIdx.x & 63;
    int rg = threadIdx.x >> 6;           // 0..3
    int col = blockIdx.x * 64 + c;
    float s = 0.f, q = 0.f;
    if (col < N) {
        for (int r = rg; r < M; r += 4) {
            float v = X[(size_t)r * stride + col];
            s += v; q += v * v;
        }
    }
    __shared__ float ls[4][64], lq[4][64];
    ls[rg][c] = s; lq[rg][c] = q;
    __syncthreads();
    if (rg == 0 && col < N) {
        s = ls[0][c] + ls[1][c] + ls[2][c] + ls[3][c];
        q = lq[0][c] + lq[1][c] + lq[2][c] + lq[3][c];
        float mu  = s / (float)M;
        float var = q / (float)M - mu * mu;
        float sc  = g[col] * rsqrtf(var + 1e-5f);
        scale[col] = sc;
        shift[col] = be[col] - mu * sc;
    }
}

// bn apply fused with bf16 conversion (pad columns -> 0)
__global__ __launch_bounds__(256) void bn_apply_bf16(
    const float* __restrict__ X, u16* __restrict__ Y,
    int total, int N, int stride,
    const float* __restrict__ scale, const float* __restrict__ shift)
{
    for (int i = blockIdx.x * 256 + threadIdx.x; i < total; i += gridDim.x * 256) {
        int c = i % stride;
        float v = 0.f;
        if (c < N) v = X[i] * scale[c] + shift[c];
        Y[i] = f2bf(v);
    }
}

__global__ __launch_bounds__(256) void bn_apply_f32(
    const float* __restrict__ X, float* __restrict__ Y,
    int total, int N, int stride,
    const float* __restrict__ scale, const float* __restrict__ shift)
{
    for (int i = blockIdx.x * 256 + threadIdx.x; i < total; i += gridDim.x * 256) {
        int c = i % stride;
        float v = 0.f;
        if (c < N) v = X[i] * scale[c] + shift[c];
        Y[i] = v;
    }
}

// Weight prep: W [K][N] fp32 row-major  ->  Wt [N][Kp] bf16 (zero K-pad)
__global__ __launch_bounds__(256) void wt_bf16(
    const float* __restrict__ W, u16* __restrict__ Wt, int K, int N, int Kp)
{
    size_t tot = (size_t)N * Kp;
    for (size_t i = (size_t)blockIdx.x * 256 + threadIdx.x; i < tot;
         i += (size_t)gridDim.x * 256) {
        int n = (int)(i / Kp), k = (int)(i % Kp);
        float v = (k < K) ? W[(size_t)k * N + n] : 0.f;
        Wt[i] = f2bf(v);
    }
}

__global__ void reduce_wpf(const float* __restrict__ Wp, float* __restrict__ out)
{
    __shared__ float r[256];
    float s = 0.f;
    for (int i = threadIdx.x; i < D0; i += 256) s += Wp[i];
    r[threadIdx.x] = s;
    __syncthreads();
    for (int st = 128; st > 0; st >>= 1) {
        if (threadIdx.x < st) r[threadIdx.x] += r[threadIdx.x + st];
        __syncthreads();
    }
    if (threadIdx.x == 0) *out = r[0];
}

// ---------------------------------------------------------------------------
// bf16 WMMA GEMM:  C[M][N] = A[M][Kp] * Bt[N][Kp]^T + bias
// 128x128 block tile, 8 waves (4x2 wave grid), 32x64 per wave, Kt=32.
// Double-buffered LDS; stage k+1 copied with GLOBAL_LOAD_ASYNC_TO_LDS_B128
// (ASYNCcnt) while stage k runs through v_wmma_f32_16x16x32_bf16.
// ---------------------------------------------------------------------------
#define TM 128
#define TN 128
#define KT 32
#define LDA 40   // KT + 8 ushorts padding -> 80B row stride, conflict-spread
#define LDB 40

__global__ __launch_bounds__(256) void gemm_bf16(
    const u16* __restrict__ A,   // [M][Kp] bf16
    const u16* __restrict__ Bt,  // [N][Kp] bf16 (pre-transposed weights)
    const float* __restrict__ bias,
    float* __restrict__ C,       // [M][N] fp32
    int M, int N, int Kp)
{
    __shared__ __align__(16) u16 lA[2][TM * LDA];
    __shared__ __align__(16) u16 lB[2][TN * LDB];

    int m0 = blockIdx.x * TM;
    int n0 = blockIdx.y * TN;
    int t = threadIdx.x;
    int lane = t & 31, w = t >> 5;
    int wm = w & 3, wn = w >> 2;       // 4x2 wave grid
    int l = lane & 15, hi = lane >> 4;

    v8f acc[2][4];
    v8f zero = {0.f, 0.f, 0.f, 0.f, 0.f, 0.f, 0.f, 0.f};
#pragma unroll
    for (int i = 0; i < 2; i++)
#pragma unroll
        for (int j = 0; j < 4; j++) acc[i][j] = zero;

    // each thread moves two 16B chunks of A and two of B per K-stage
    int r0 = t >> 2,           c0 = (t & 3) * 8;         // chunk ids 0..255
    int r1 = (t + 256) >> 2,   c1 = ((t + 256) & 3) * 8; // chunk ids 256..511

    const u16* gA0 = A  + (size_t)(m0 + r0) * Kp + c0;
    const u16* gA1 = A  + (size_t)(m0 + r1) * Kp + c1;
    const u16* gB0 = Bt + (size_t)(n0 + r0) * Kp + c0;
    const u16* gB1 = Bt + (size_t)(n0 + r1) * Kp + c1;

    auto issue = [&](int buf, int k0) {
        cp16(gA0 + k0, &lA[buf][r0 * LDA + c0]);
        cp16(gA1 + k0, &lA[buf][r1 * LDA + c1]);
        cp16(gB0 + k0, &lB[buf][r0 * LDB + c0]);
        cp16(gB1 + k0, &lB[buf][r1 * LDB + c1]);
    };

    int nk = Kp / KT;
    issue(0, 0);                               // prologue: stage 0

    for (int ks = 0; ks < nk; ks++) {
        int buf = ks & 1;
        if (ks + 1 < nk) {
            __syncthreads();                   // all waves done reading buf^1
            issue(buf ^ 1, (ks + 1) * KT);     // overlap next copy w/ this compute
            if (ks + 2 < nk) {                 // deep prefetch of stage k+2
                __builtin_prefetch((const void*)(gA0 + (ks + 2) * KT), 0, 3);
                __builtin_prefetch((const void*)(gB0 + (ks + 2) * KT), 0, 3);
            }
            WAIT_ASYNC(4);                     // stage ks landed (4 older ops)
        } else {
            WAIT_ASYNC(0);
        }
        __syncthreads();                       // stage ks visible block-wide

        union Frag { u32x4 q[2]; v16bf v; } fa[2], fb[4];
#pragma unroll
        for (int i = 0; i < 2; i++) {
            const u16* p = &lA[buf][(wm * 32 + i * 16 + l) * LDA];
            fa[i].q[0] = *(const u32x4*)(p + hi * 8);        // K 0..7 / 8..15
            fa[i].q[1] = *(const u32x4*)(p + 16 + hi * 8);   // K 16..23 / 24..31
        }
#pragma unroll
        for (int j = 0; j < 4; j++) {
            const u16* p = &lB[buf][(wn * 64 + j * 16 + l) * LDB];
            fb[j].q[0] = *(const u32x4*)(p + hi * 8);
            fb[j].q[1] = *(const u32x4*)(p + 16 + hi * 8);
        }
#pragma unroll
        for (int i = 0; i < 2; i++)
#pragma unroll
            for (int j = 0; j < 4; j++)
                acc[i][j] = __builtin_amdgcn_wmma_f32_16x16x32_bf16(
                    false, fa[i].v, false, fb[j].v,
                    (short)0, acc[i][j], false, false);
    }

    // epilogue: C/D layout -> VGPR r: lanes0-15 M=r, lanes16-31 M=r+8; N=lane&15
#pragma unroll
    for (int i = 0; i < 2; i++) {
#pragma unroll
        for (int j = 0; j < 4; j++) {
            int col = n0 + wn * 64 + j * 16 + l;
            float bv = bias[col];
#pragma unroll
            for (int r = 0; r < 8; r++) {
                int row = m0 + wm * 32 + i * 16 + hi * 8 + r;
                C[(size_t)row * N + col] = acc[i][j][r] + bv;
            }
        }
    }
}

// ---------------------------------------------------------------------------
// Final: out[b] = sigmoid((1+s)*p + bc2*sum(Wp_front) + h3[b]·Wp_back + bp)
// one wave per row
// ---------------------------------------------------------------------------
__global__ __launch_bounds__(256) void final_out(
    const float* __restrict__ h3, const float* __restrict__ s_arr,
    const float* __restrict__ p_arr, const float* __restrict__ Wp,
    const float* __restrict__ bp, const float* __restrict__ b_cross,
    const float* __restrict__ sumwpf, float* __restrict__ out)
{
    int w = threadIdx.x >> 5, lane = threadIdx.x & 31;
    int b = blockIdx.x * 8 + w;
    const float* Wpb = Wp + D0;
    float d = 0.f;
    for (int k = lane; k < H3; k += 32) d += h3[(size_t)b * H3 + k] * Wpb[k];
#pragma unroll
    for (int m = 16; m; m >>= 1) d += __shfl_xor(d, m, 32);
    if (lane == 0) {
        float v = (1.f + s_arr[b]) * p_arr[b] + b_cross[2] * (*sumwpf) + d + bp[0];
        out[b] = 1.f / (1.f + __expf(-v));
    }
}

// ---------------------------------------------------------------------------
extern "C" void kernel_launch(void* const* d_in, const int* in_sizes, int n_in,
                              void* d_out, int out_size, void* d_ws, size_t ws_size,
                              hipStream_t stream)
{
    const int*   sparse  = (const int*)  d_in[0];
    const float* dense   = (const float*)d_in[1];
    const float* emb     = (const float*)d_in[2];
    const float* w_cross = (const float*)d_in[3];
    const float* b_cross = (const float*)d_in[4];
    const float* bn0g    = (const float*)d_in[5];
    const float* bn0b    = (const float*)d_in[6];
    const float* W1      = (const float*)d_in[7];
    const float* bias1   = (const float*)d_in[8];
    const float* bn1g    = (const float*)d_in[9];
    const float* bn1b    = (const float*)d_in[10];
    const float* W2      = (const float*)d_in[11];
    const float* bias2   = (const float*)d_in[12];
    const float* bn2g    = (const float*)d_in[13];
    const float* bn2b    = (const float*)d_in[14];
    const float* W3      = (const float*)d_in[15];
    const float* bias3   = (const float*)d_in[16];
    const float* bn3g    = (const float*)d_in[17];
    const float* bn3b    = (const float*)d_in[18];
    const float* Wp      = (const float*)d_in[19];
    const float* bp      = (const float*)d_in[20];

    char* ws = (char*)d_ws;
    // regionA (fp32, BATCH*D0P): x0, later c1/c2/c3   (111.1 MB)
    // regionB (bf16, BATCH*D0P): h0, later h1/h2, finally h3 fp32 (55.6 MB)
    size_t offA    = 0;
    size_t offB    = offA    + (size_t)BATCH * D0P * 4;
    size_t offWt1  = offB    + (size_t)BATCH * D0P * 2;
    size_t offWt2  = offWt1  + (size_t)H1 * D0P * 2;
    size_t offWt3  = offWt2  + (size_t)H2 * H1  * 2;
    size_t offScal = offWt3  + (size_t)H3 * H2  * 2;
    size_t offShft = offScal + (size_t)D0P * 4;
    size_t offS    = offShft + (size_t)D0P * 4;
    size_t offP    = offS    + (size_t)BATCH * 4;
    size_t offSum  = offP    + (size_t)BATCH * 4;

    float* x0    = (float*)(ws + offA);
    float* cbuf  = (float*)(ws + offA);   // reuses x0 region after h0 is made
    u16*   hbuf  = (u16*)  (ws + offB);   // h0/h1/h2 bf16
    float* h3    = (float*)(ws + offB);   // reuses regionB after GEMM3
    u16*   Wt1   = (u16*)  (ws + offWt1);
    u16*   Wt2   = (u16*)  (ws + offWt2);
    u16*   Wt3   = (u16*)  (ws + offWt3);
    float* scale = (float*)(ws + offScal);
    float* shift = (float*)(ws + offShft);
    float* s_arr = (float*)(ws + offS);
    float* p_arr = (float*)(ws + offP);
    float* swpf  = (float*)(ws + offSum);
    float* out   = (float*)d_out;

    // weight prep + projection-sum (tiny, one-time per call)
    reduce_wpf<<<1, 256, 0, stream>>>(Wp, swpf);
    wt_bf16<<<1024, 256, 0, stream>>>(W1, Wt1, D0, H1, D0P);
    wt_bf16<<<512,  256, 0, stream>>>(W2, Wt2, H1, H2, H1);
    wt_bf16<<<256,  256, 0, stream>>>(W3, Wt3, H2, H3, H2);

    // gather + cross/projection row dots
    gather_x0<<<BATCH, 256, 0, stream>>>(sparse, dense, emb,
                                         w_cross + 2 * D0, Wp, x0, s_arr, p_arr);

    // bn0 -> h0 bf16
    col_stats<<<(D0 + 63) / 64, 256, 0, stream>>>(x0, BATCH, D0, D0P, bn0g, bn0b, scale, shift);
    bn_apply_bf16<<<4096, 256, 0, stream>>>(x0, hbuf, BATCH * D0P, D0, D0P, scale, shift);

    // layer 1
    gemm_bf16<<<dim3(BATCH / TM, H1 / TN), 256, 0, stream>>>(hbuf, Wt1, bias1, cbuf, BATCH, H1, D0P);
    col_stats<<<H1 / 64, 256, 0, stream>>>(cbuf, BATCH, H1, H1, bn1g, bn1b, scale, shift);
    bn_apply_bf16<<<4096, 256, 0, stream>>>(cbuf, hbuf, BATCH * H1, H1, H1, scale, shift);

    // layer 2
    gemm_bf16<<<dim3(BATCH / TM, H2 / TN), 256, 0, stream>>>(hbuf, Wt2, bias2, cbuf, BATCH, H2, H1);
    col_stats<<<H2 / 64, 256, 0, stream>>>(cbuf, BATCH, H2, H2, bn2g, bn2b, scale, shift);
    bn_apply_bf16<<<4096, 256, 0, stream>>>(cbuf, hbuf, BATCH * H2, H2, H2, scale, shift);

    // layer 3
    gemm_bf16<<<dim3(BATCH / TM, H3 / TN), 256, 0, stream>>>(hbuf, Wt3, bias3, cbuf, BATCH, H3, H2);
    col_stats<<<H3 / 64, 256, 0, stream>>>(cbuf, BATCH, H3, H3, bn3g, bn3b, scale, shift);
    bn_apply_f32<<<4096, 256, 0, stream>>>(cbuf, h3, BATCH * H3, H3, H3, scale, shift);

    // final projection + sigmoid
    final_out<<<BATCH / 8, 256, 0, stream>>>(h3, s_arr, p_arr, Wp, bp, b_cross, swpf, out);
}
